// TransformerEncoder_7533372637684
// MI455X (gfx1250) — matrix operand order repaired and checked by reading, not compiled, not run on hardware
//
#include <hip/hip_runtime.h>
#include <hip/hip_bf16.h>
#include <stddef.h>

typedef __attribute__((ext_vector_type(16))) __bf16        v16bf;
typedef __attribute__((ext_vector_type(8)))  float         v8f;
typedef __attribute__((ext_vector_type(4)))  float         v4f;
typedef __attribute__((ext_vector_type(8)))  unsigned int  v8u;

// ---------------------------------------------------------------------------
// helpers
// ---------------------------------------------------------------------------
static __device__ __forceinline__ unsigned short f2bfu(float f) {
    // round-to-nearest-even fp32 -> bf16 (bit pattern as ushort)
    unsigned int u = __builtin_bit_cast(unsigned int, f);
    unsigned int r = u + 0x7FFFu + ((u >> 16) & 1u);
    return (unsigned short)(r >> 16);
}

// ---------------------------------------------------------------------------
// Kernel 1: x = feats @ W + b   (fp32 in, bf16 WMMA, fp32 accumulate/out)
// Block tile 128(M) x 64(N), BK=64. 256 threads = 8 wave32 (4x2 waves),
// each wave computes 32x32 as 2x2 WMMA tiles over 2 k-subtiles = 8 wmma/iter.
// Global loads batched into registers before LDS staging (MLP), coalesced
// float4 epilogue through LDS with fused bias.
// ---------------------------------------------------------------------------
#define BM 128
#define BN 64
#define BK 64
#define LDSK (BK + 2)   // even element stride -> 4B-aligned 32b LDS reads

#define A_BYTES (BM * LDSK * 2)
#define B_BYTES (BN * LDSK * 2)
#define C_BYTES (BM * BN * 4)
#define SMEM_BYTES (C_BYTES > (A_BYTES + B_BYTES) ? C_BYTES : (A_BYTES + B_BYTES))

__global__ __launch_bounds__(256)
void gemm_bias_wmma_bf16(const float* __restrict__ A,    // (M,K) feats
                         const float* __restrict__ W,    // (K,N)
                         const float* __restrict__ bias, // (N)
                         float* __restrict__ C,          // (M,N)
                         int M, int N, int K) {
    __shared__ __align__(16) unsigned char smem[SMEM_BYTES];
    unsigned short (*As)[LDSK] = (unsigned short (*)[LDSK])smem;             // [BM][LDSK]
    unsigned short (*Bs)[LDSK] = (unsigned short (*)[LDSK])(smem + A_BYTES); // [BN][LDSK]

    const int tid   = threadIdx.x;
    const int mb0   = blockIdx.y * BM;
    const int nb0   = blockIdx.x * BN;
    const int wave  = tid >> 5;
    const int lane  = tid & 31;
    const int wm    = (wave & 3) * 32;  // wave M offset in block tile
    const int wn    = (wave >> 2) * 32; // wave N offset in block tile
    const int mlane = lane & 15;
    const int hi    = lane >> 4;        // 0: lanes 0-15, 1: lanes 16-31

    v8f acc[2][2];
    const v8f zero8 = {};
    for (int i = 0; i < 2; ++i)
        for (int j = 0; j < 2; ++j)
            acc[i][j] = zero8;

    // staging-chunk coordinates (constant across k-loop)
    // A: 128x64 fp32 = 2048 float4, 8 per thread; B: 64x64 = 1024 float4, 4/thr
    int arow[8], acol[8], bk[4], bn4[4];
    #pragma unroll
    for (int it = 0; it < 8; ++it) {
        int c = tid + 256 * it;
        arow[it] = c >> 4;            // 16 float4 per 64-wide row
        acol[it] = (c & 15) * 4;
    }
    #pragma unroll
    for (int it = 0; it < 4; ++it) {
        int c = tid + 256 * it;
        bk[it]  = c >> 4;
        bn4[it] = (c & 15) * 4;
    }

    for (int kt = 0; kt < K; kt += BK) {
        // ---- batch ALL global loads into registers (12 outstanding) ----
        v4f areg[8], breg[4];
        #pragma unroll
        for (int it = 0; it < 8; ++it)
            areg[it] = *(const v4f*)(A + (size_t)(mb0 + arow[it]) * K + kt + acol[it]);
        #pragma unroll
        for (int it = 0; it < 4; ++it)
            breg[it] = *(const v4f*)(W + (size_t)(kt + bk[it]) * N + nb0 + bn4[it]);
        if (kt + BK < K)  // global_prefetch for next A tile
            __builtin_prefetch(A + (size_t)(mb0 + (tid >> 1)) * K + kt + BK, 0, 0);

        __syncthreads();  // previous iteration done reading LDS

        // ---- convert fp32 -> bf16 and stage to LDS ----
        #pragma unroll
        for (int it = 0; it < 8; ++it) {
            As[arow[it]][acol[it] + 0] = f2bfu(areg[it].x);
            As[arow[it]][acol[it] + 1] = f2bfu(areg[it].y);
            As[arow[it]][acol[it] + 2] = f2bfu(areg[it].z);
            As[arow[it]][acol[it] + 3] = f2bfu(areg[it].w);
        }
        #pragma unroll
        for (int it = 0; it < 4; ++it) {   // transpose W tile: Bs[n][k]
            Bs[bn4[it] + 0][bk[it]] = f2bfu(breg[it].x);
            Bs[bn4[it] + 1][bk[it]] = f2bfu(breg[it].y);
            Bs[bn4[it] + 2][bk[it]] = f2bfu(breg[it].z);
            Bs[bn4[it] + 3][bk[it]] = f2bfu(breg[it].w);
        }
        __syncthreads();

        // ---- 2 k-subtiles of 32: load fragments, 4 wmma each ----
        #pragma unroll
        for (int q = 0; q < 2; ++q) {
            v8u afr[2], bfr[2];
            #pragma unroll
            for (int s = 0; s < 2; ++s) {   // A 16x32 fragment (ISA layout)
                int m  = wm + s * 16 + mlane;
                int kb = q * 32 + hi * 8;   // lanes 0-15: K{0-7,16-23}; hi: +8
                #pragma unroll
                for (int v = 0; v < 4; ++v) {
                    afr[s][v]     = *(const unsigned int*)&As[m][kb + 2 * v];
                    afr[s][4 + v] = *(const unsigned int*)&As[m][16 + kb + 2 * v];
                }
            }
            #pragma unroll
            for (int s = 0; s < 2; ++s) {   // B 32x16 fragment (ISA layout)
                int n    = wn + s * 16 + mlane;
                int koff = q * 32 + hi * 8;  // lanes 0-15: K{0-7,16-23}; hi: +8
                #pragma unroll
                for (int v = 0; v < 8; ++v)
                    bfr[s][v] = *(const unsigned int*)&Bs[n][koff + 2 * v + (v >= 4 ? 8 : 0)];
            }
            #pragma unroll
            for (int i = 0; i < 2; ++i)
                #pragma unroll
                for (int j = 0; j < 2; ++j)
                    acc[i][j] = __builtin_amdgcn_wmma_f32_16x16x32_bf16(
                        false, __builtin_bit_cast(v16bf, afr[i]),
                        false, __builtin_bit_cast(v16bf, bfr[j]),
                        (short)0, acc[i][j], false, false);
        }
    }

    // ---- epilogue: stage C tile through LDS, coalesced float4 writeback ----
    __syncthreads();                       // all LDS fragment reads done
    float (*Cs)[BN] = (float (*)[BN])smem; // reuse 32KB
    #pragma unroll
    for (int i = 0; i < 2; ++i) {
        int rowBase = wm + i * 16 + hi * 8; // VGPR r -> M=r (+8 for hi lanes)
        #pragma unroll
        for (int j = 0; j < 2; ++j) {
            int col = wn + j * 16 + mlane;  // N = lane&15
            #pragma unroll
            for (int r = 0; r < 8; ++r)
                Cs[rowBase + r][col] = acc[i][j][r];
        }
    }
    __syncthreads();
    #pragma unroll
    for (int it = 0; it < 8; ++it) {
        int c    = tid + 256 * it;          // 2048 float4 of the 128x64 tile
        int row  = c >> 4;
        int col4 = (c & 15) * 4;
        v4f v  = *(const v4f*)&Cs[row][col4];
        v4f bv = *(const v4f*)(bias + nb0 + col4);
        v += bv;
        *(v4f*)(C + (size_t)(mb0 + row) * N + nb0 + col4) = v;
    }
}

// ---------------------------------------------------------------------------
// Kernel 2: per-row fused scans (16 rows). One 256-thread block per batch row,
// 8 contiguous elements per thread. Inclusive cumsum of mask (position_ids)
// and inclusive cummax of (mask ? 0 : j) (episode start) in a single pass.
// ---------------------------------------------------------------------------
__global__ __launch_bounds__(256)
void scan_kernel(const float* __restrict__ masks, // (B*S)
                 float* __restrict__ pos_out,     // (B*S)
                 int* __restrict__ ep_out,        // (B*S)
                 int S) {
    const int b    = blockIdx.x;
    const int t    = threadIdx.x;
    const int PER  = S / 256;                // 8
    const int base = b * S + t * PER;

    float lm[8], csum[8];
    int   cmax[8];
    float s  = 0.f;
    int   mx = 0;
    for (int i = 0; i < PER; ++i) {
        float mf = (masks[base + i] != 0.f) ? 1.f : 0.f;
        lm[i] = mf;
        s += mf;
        csum[i] = s;
        int cand = (mf != 0.f) ? 0 : (t * PER + i);
        mx = max(mx, cand);
        cmax[i] = mx;
    }

    __shared__ float ssum[256];
    __shared__ int   smax[256];
    ssum[t] = s;
    smax[t] = mx;
    __syncthreads();
    for (int off = 1; off < 256; off <<= 1) {
        float vs = 0.f;
        int   vm = 0;
        if (t >= off) { vs = ssum[t - off]; vm = smax[t - off]; }
        __syncthreads();
        ssum[t] += vs;
        smax[t] = max(smax[t], vm);
        __syncthreads();
    }
    float psum = (t > 0) ? ssum[t - 1] : 0.f;
    int   pmax = (t > 0) ? smax[t - 1] : 0;

    for (int i = 0; i < PER; ++i) {
        float p = psum + csum[i];
        pos_out[base + i] = (lm[i] != 0.f) ? p : 0.f;
        ep_out[base + i]  = max(pmax, cmax[i]);
    }
}

// ---------------------------------------------------------------------------
// Kernel 3: attn_mask fill (B,S,S). One block per (b,i) row; non-temporal
// v4f streaming stores (268 MB can't fit L2; don't pollute it).
// allowed(j) = ep_start <= j <= i.
// ---------------------------------------------------------------------------
__global__ __launch_bounds__(256)
void attn_mask_kernel(const int* __restrict__ ep,  // (B*S)
                      float* __restrict__ out,     // (B*S*S)
                      int S) {
    const size_t row = blockIdx.x;               // 0 .. B*S-1
    const int    i   = (int)(row & (size_t)(S - 1));
    const int    e   = ep[row];
    v4f* orow = (v4f*)(out + row * (size_t)S);
    const float NEGINF = -__builtin_inff();

    for (int j4 = threadIdx.x; j4 < (S >> 2); j4 += 256) {
        int j = j4 * 4;
        v4f v;
        v.x = (j + 0 >= e && j + 0 <= i) ? 0.f : NEGINF;
        v.y = (j + 1 >= e && j + 1 <= i) ? 0.f : NEGINF;
        v.z = (j + 2 >= e && j + 2 <= i) ? 0.f : NEGINF;
        v.w = (j + 3 >= e && j + 3 <= i) ? 0.f : NEGINF;
        __builtin_nontemporal_store(v, &orow[j4]);
    }
}

// ---------------------------------------------------------------------------
// launch
// ---------------------------------------------------------------------------
extern "C" void kernel_launch(void* const* d_in, const int* in_sizes, int n_in,
                              void* d_out, int out_size, void* d_ws, size_t ws_size,
                              hipStream_t stream) {
    const float* feats = (const float*)d_in[0];   // (B*S, D)
    const float* masks = (const float*)d_in[1];   // (B*S, 1)
    const float* W     = (const float*)d_in[2];   // (D, E)
    const float* bias  = (const float*)d_in[3];   // (E)

    const int S  = 2048;                          // num_steps (reference)
    const int BS = in_sizes[1];                   // B*S = 32768
    const int B  = BS / S;                        // 16
    const int D  = in_sizes[0] / BS;              // 1024
    const int E  = in_sizes[3];                   // 1024

    float* x_out   = (float*)d_out;                               // B*S*E
    float* pos_out = x_out + (size_t)BS * E;                      // B*S
    float* am_out  = pos_out + BS;                                // B*S*S
    int*   ep_ws   = (int*)d_ws;                                  // B*S ints

    // GEMM: M=BS, N=E, K=D
    dim3 ggrid(E / BN, BS / BM);
    gemm_bias_wmma_bf16<<<ggrid, 256, 0, stream>>>(feats, W, bias, x_out,
                                                   BS, E, D);

    // fused scans: one block per batch row
    scan_kernel<<<B, 256, 0, stream>>>(masks, pos_out, ep_ws, S);

    // attention mask fill: one block per (b,i) row
    attn_mask_kernel<<<(unsigned)BS, 256, 0, stream>>>(ep_ws, am_out, S);
}
